// SSMTokenMixer_53592601919829
// MI455X (gfx1250) — matrix operand order repaired and checked by reading, not compiled, not run on hardware
//
#include <hip/hip_runtime.h>

typedef __attribute__((ext_vector_type(16))) __bf16 v16bf;
typedef __attribute__((ext_vector_type(8)))  float  v8f;

// Problem constants
#define NB   16
#define CH   128
#define HWSZ 4096          // H*W
#define TLEN 64
#define G3   384           // 3*C

// Workspace layout (bytes)
static constexpr size_t XN_OFF  = 0;                 // bf16 [1024][64][128] = 16 MB
static constexpr size_t XNT_OFF = 16777216;          // bf16 transposed copy = 16 MB
static constexpr size_t W_OFF   = 33554432;          // bf16 weights 4x(wih+whh) = 768 KB
static constexpr size_t B_OFF   = 34340864;          // f32 biases 4x(bih+bhh)   = 12 KB
static constexpr size_t PW_OFF  = 34353152;          // bf16 proj_w [128][512]   = 128 KB
static constexpr size_t GO_OFF  = 34484224;          // bf16 gru out [65536][512] = 64 MB

struct InPtrs {
  const float* x; const float* g; const float* b;
  const float* wih[4]; const float* whh[4];
  const float* bih[4]; const float* bhh[4];
  const float* pw; const float* pb;
};

// ---------------- prep: convert weights/biases to ws ----------------
__global__ void prep_kernel(InPtrs p, __bf16* __restrict__ wdst,
                            float* __restrict__ bdst, __bf16* __restrict__ pwdst) {
  int tid = blockIdx.x * 256 + threadIdx.x;
  if (tid < 4 * 2 * G3 * CH) {                       // 393216 weight elems
    int combo = tid / (2 * G3 * CH);
    int r     = tid % (2 * G3 * CH);
    const float* src = (r < G3 * CH) ? p.wih[combo] : p.whh[combo];
    wdst[tid] = (__bf16)src[r % (G3 * CH)];
  }
  if (tid < CH * 4 * CH) {                           // proj_w 128x512
    pwdst[tid] = (__bf16)p.pw[tid];
  }
  if (tid < 4 * 2 * G3) {                            // biases
    int combo = tid / (2 * G3);
    int r     = tid % (2 * G3);
    bdst[tid] = (r < G3) ? p.bih[combo][r] : p.bhh[combo][r - G3];
  }
}

// ---------------- LayerNorm over C, emit bf16 in two layouts ----------------
__global__ void __launch_bounds__(256) ln_kernel(
    const float* __restrict__ x, const float* __restrict__ g,
    const float* __restrict__ bb, __bf16* __restrict__ xn,
    __bf16* __restrict__ xnt) {
  int wv = threadIdx.x >> 5, lane = threadIdx.x & 31;
  int pos = blockIdx.x * 8 + wv;                     // 0..65535
  int b = pos >> 12, hw = pos & 4095;
  int hrow = hw >> 6, wcol = hw & 63;
  float v[4]; float s = 0.f, s2 = 0.f;
  for (int j = 0; j < 4; ++j) {
    int c = lane + j * 32;
    float t = x[((size_t)b * CH + c) * HWSZ + hw];
    v[j] = t; s += t; s2 += t * t;
  }
  for (int m = 16; m >= 1; m >>= 1) {
    s  += __shfl_xor(s,  m, 32);
    s2 += __shfl_xor(s2, m, 32);
  }
  float mu  = s * (1.f / 128.f);
  float var = s2 * (1.f / 128.f) - mu * mu;
  float rs  = rsqrtf(var + 1e-5f);
  size_t o1 = (size_t)pos * CH;
  size_t o2 = ((size_t)b * HWSZ + wcol * 64 + hrow) * CH;
  for (int j = 0; j < 4; ++j) {
    int c = lane + j * 32;
    float y = (v[j] - mu) * rs * g[c] + bb[c];
    __bf16 yb = (__bf16)y;
    xn[o1 + c]  = yb;
    xnt[o2 + c] = yb;
  }
}

// ---------------- WMMA / math helpers ----------------
__device__ __forceinline__ v16bf load_a16(const __bf16* Ls, int lane, int kc, int stride) {
  // A 16x32 load, contiguous k (B matches): lane 0-15 -> M=lane, K base 0; lane 16-31 -> +16
  // halves 0..15 -> K=kb..kb+15 (two 16B runs)
  int m  = lane & 15;
  int kb = (lane >> 4) << 4;
  int base = m * stride + kc * 32 + kb;
  union { v16bf v; uint4 q[2]; } u;
  u.q[0] = *(const uint4*)(Ls + base);
  u.q[1] = *(const uint4*)(Ls + base + 8);
  return u.v;
}

__device__ __forceinline__ v8f wmma_bf16(v16bf a, v16bf b, v8f c) {
  return __builtin_amdgcn_wmma_f32_16x16x32_bf16(false, a, false, b, (short)0, c,
                                                 false, false);
}

// v_rcp_f32 + v_exp_f32 based activations: avoid IEEE-div expansion and libm
// tanh on the serial recurrence path.
__device__ __forceinline__ float fast_sigmoid(float x) {
  return __builtin_amdgcn_rcpf(1.f + __expf(-x));
}
__device__ __forceinline__ float fast_tanh(float x) {
  return 2.f * __builtin_amdgcn_rcpf(1.f + __expf(-2.f * x)) - 1.f;
}

// ---------------- fused bidirectional GRU scan ----------------
// grid (64, 4): x = sequence block of 16, y = combo {h_f, h_b, v_f, v_b}
__global__ void __launch_bounds__(256, 1) gru_kernel(
    const __bf16* __restrict__ xn, const __bf16* __restrict__ xnt,
    const __bf16* __restrict__ wts, const float* __restrict__ biases,
    __bf16* __restrict__ gout) {
  const int combo    = blockIdx.y;
  const int reverse  = combo & 1;
  const int vertical = combo >> 1;
  const __bf16* xin = vertical ? xnt : xn;
  const __bf16* wih = wts + (size_t)combo * (2 * G3 * CH);
  const __bf16* whh = wih + (size_t)G3 * CH;
  const float*  bih = biases + combo * (2 * G3);
  const float*  bhh = bih + G3;
  const int s0   = blockIdx.x * 16;
  const int tid  = threadIdx.x, wv = tid >> 5, lane = tid & 31;

  __shared__ __bf16 Xs[16 * 136];
  __shared__ __bf16 Hs[16 * 136];

  // Pin W_ih / W_hh B-fragments in VGPRs. B(k,n) = W[n][k]; per lane the 16
  // halves are K-contiguous -> one 32B load each.
  v16bf bIH[3][4], bHH[3][4];
  {
    int nloc = lane & 15, khalf = (lane >> 4) << 4;
    for (int gi = 0; gi < 3; ++gi) {
      int nrow = (gi * 8 + wv) * 16 + nloc;          // gate-tile row in [0,384)
      for (int kc = 0; kc < 4; ++kc) {
        bIH[gi][kc] = *(const v16bf*)(wih + (size_t)nrow * CH + kc * 32 + khalf);
        bHH[gi][kc] = *(const v16bf*)(whh + (size_t)nrow * CH + kc * 32 + khalf);
      }
    }
  }
  const int ch  = wv * 16 + (lane & 15);             // channel this lane owns
  const float br  = bih[ch]          + bhh[ch];      // folded r/z biases
  const float bz  = bih[CH + ch]     + bhh[CH + ch];
  const float bni = bih[2 * CH + ch];
  const float bnh = bhh[2 * CH + ch];

  for (int i = tid; i < 16 * 136; i += 256) Hs[i] = (__bf16)0.f;
  float hreg[8];
  for (int i = 0; i < 8; ++i) hreg[i] = 0.f;

  // Per-thread X staging address (constant row/col across t)
  const int xrow = tid >> 4;                          // (tid*8)>>7
  const int xcol = (tid & 15) << 3;                   // (tid*8)&127
  const size_t xbase = (size_t)(s0 + xrow) * TLEN * CH + xcol;
  __bf16* xs_dst = Xs + xrow * 136 + xcol;

  // software pipeline: prefetch X(t=0)
  int teff = reverse ? (TLEN - 1) : 0;
  uint4 qx = *(const uint4*)(xin + xbase + (size_t)teff * CH);

  __syncthreads();

  for (int t = 0; t < TLEN; ++t) {
    teff = reverse ? (TLEN - 1 - t) : t;
    *(uint4*)xs_dst = qx;                             // stage X(t)
    __syncthreads();

    if (t + 1 < TLEN) {                               // prefetch X(t+1): latency
      int tn = reverse ? (TLEN - 2 - t) : (t + 1);    // hides under WMMA phase
      qx = *(const uint4*)(xin + xbase + (size_t)tn * CH);
    }

    v8f aXr = {}, aXz = {}, aXn = {}, aHr = {}, aHz = {}, aHn = {};
    for (int kc = 0; kc < 4; ++kc) {
      v16bf ax = load_a16(Xs, lane, kc, 136);
      v16bf ah = load_a16(Hs, lane, kc, 136);
      aXr = wmma_bf16(ax, bIH[0][kc], aXr);
      aXz = wmma_bf16(ax, bIH[1][kc], aXz);
      aXn = wmma_bf16(ax, bIH[2][kc], aXn);
      aHr = wmma_bf16(ah, bHH[0][kc], aHr);
      aHz = wmma_bf16(ah, bHH[1][kc], aHz);
      aHn = wmma_bf16(ah, bHH[2][kc], aHn);
    }
    // GRU gate math; lane-local thanks to {w, w+8, w+16} tile assignment
    for (int i = 0; i < 8; ++i) {
      float r = fast_sigmoid(aXr[i] + aHr[i] + br);
      float z = fast_sigmoid(aXz[i] + aHz[i] + bz);
      float n = fast_tanh(aXn[i] + bni + r * (aHn[i] + bnh));
      hreg[i] = n + z * (hreg[i] - n);
    }
    __syncthreads();                                 // all Hs reads complete
    for (int i = 0; i < 8; ++i) {
      int m = i + ((lane >> 4) << 3);                // row per C/D layout
      __bf16 hb = (__bf16)hreg[i];
      Hs[m * 136 + ch] = hb;
      int seq = s0 + m;
      int pos = vertical ? (((seq >> 6) << 12) + (seq & 63) + (teff << 6))
                         : ((seq << 6) + teff);
      gout[(size_t)pos * 512 + combo * CH + ch] = hb;
    }
    __syncthreads();
  }
}

// ---------------- projection + bias + residual, transposed output ----------------
__global__ void __launch_bounds__(256, 1) proj_kernel(
    const __bf16* __restrict__ gin, const __bf16* __restrict__ pw,
    const float* __restrict__ pb, const float* __restrict__ x,
    float* __restrict__ out) {
  const int tid = threadIdx.x, wv = tid >> 5, lane = tid & 31;
  __shared__ __bf16 Ls[16 * 520];
  __shared__ float  Ts[128 * 17];

  v16bf bPW[16];                                     // proj_w fragments pinned
  {
    int c = wv * 16 + (lane & 15);
    int khalf = (lane >> 4) << 4;
    for (int kc = 0; kc < 16; ++kc)
      bPW[kc] = *(const v16bf*)(pw + (size_t)c * 512 + kc * 32 + khalf);
  }
  for (int it = 0; it < 16; ++it) {
    const int mt   = blockIdx.x + it * 256;          // 4096 M-tiles total
    const int pos0 = mt * 16;
    // stage A [16][512] bf16
    for (int q = 0; q < 4; ++q) {
      int e = tid + q * 256;                         // uint4 index, 1024 total
      int row = e >> 6, col = (e & 63) << 3;
      const uint4 d = *(const uint4*)(gin + (size_t)(pos0 + row) * 512 + col);
      *(uint4*)(Ls + row * 520 + col) = d;
    }
    __syncthreads();
    v8f acc = {};
    for (int kc = 0; kc < 16; ++kc) {
      v16bf a = load_a16(Ls, lane, kc, 520);
      acc = wmma_bf16(a, bPW[kc], acc);
    }
    {
      int c = wv * 16 + (lane & 15);
      for (int i = 0; i < 8; ++i) {
        int m = i + ((lane >> 4) << 3);
        Ts[c * 17 + m] = acc[i];                     // LDS transpose
      }
    }
    __syncthreads();
    {
      int c  = tid >> 1;
      int m0 = (tid & 1) << 3;
      int b  = pos0 >> 12, hw0 = pos0 & 4095;
      float bias = pb[c];
      size_t obase = ((size_t)(b * CH + c)) * HWSZ + hw0 + m0;
      for (int j = 0; j < 8; ++j)                    // 32B coalesced runs
        out[obase + j] = Ts[c * 17 + m0 + j] + bias + x[obase + j];
    }
  }
}

extern "C" void kernel_launch(void* const* d_in, const int* in_sizes, int n_in,
                              void* d_out, int out_size, void* d_ws, size_t ws_size,
                              hipStream_t stream) {
  (void)in_sizes; (void)n_in; (void)out_size; (void)ws_size;
  char* ws = (char*)d_ws;
  __bf16* xn    = (__bf16*)(ws + XN_OFF);
  __bf16* xnt   = (__bf16*)(ws + XNT_OFF);
  __bf16* wdst  = (__bf16*)(ws + W_OFF);
  float*  bdst  = (float*)(ws + B_OFF);
  __bf16* pwdst = (__bf16*)(ws + PW_OFF);
  __bf16* gout  = (__bf16*)(ws + GO_OFF);

  InPtrs p;
  p.x = (const float*)d_in[0];
  p.g = (const float*)d_in[1];
  p.b = (const float*)d_in[2];
  for (int c = 0; c < 4; ++c) {                      // h_f, h_b, v_f, v_b
    p.wih[c] = (const float*)d_in[3 + c * 4 + 0];
    p.whh[c] = (const float*)d_in[3 + c * 4 + 1];
    p.bih[c] = (const float*)d_in[3 + c * 4 + 2];
    p.bhh[c] = (const float*)d_in[3 + c * 4 + 3];
  }
  p.pw = (const float*)d_in[19];
  p.pb = (const float*)d_in[20];

  prep_kernel<<<1536, 256, 0, stream>>>(p, wdst, bdst, pwdst);
  ln_kernel<<<8192, 256, 0, stream>>>(p.x, p.g, p.b, xn, xnt);
  gru_kernel<<<dim3(64, 4), 256, 0, stream>>>(xn, xnt, wdst, bdst, gout);
  proj_kernel<<<256, 256, 0, stream>>>(gout, pwdst, p.pb, p.x, (float*)d_out);
}
